// SDRProjection_42623255445785
// MI455X (gfx1250) — matrix operand order repaired and checked by reading, not compiled, not run on hardware
//
#include <hip/hip_runtime.h>
#include <hip/hip_bf16.h>

// Problem sizes (fixed by the reference)
#define BATCH_S 16384   // B*S = 4*4096
#define SDRN    2048
#define HID     2048
#define NW      41
#define EPS     1e-5f

// GEMM tiling
#define MB 128
#define NB 128
#define KC 32
#define KSTEPS (SDRN / KC)   // 64

typedef __attribute__((ext_vector_type(16))) _Float16 v16h;
typedef __attribute__((ext_vector_type(8)))  float    v8f;
typedef __attribute__((ext_vector_type(4)))  unsigned v4u;

union FragU {
    unsigned int u[8];
    v4u  q[2];
    v16h h;
};

// ---------------------------------------------------------------------------
// Kernel 1: transpose + f32->f16 convert  proj_w[H][N] -> Wt[N][H]
// ---------------------------------------------------------------------------
__global__ __launch_bounds__(256)
void sdr_transpose_kernel(const float* __restrict__ proj_w,
                          _Float16* __restrict__ Wt) {
    __shared__ float t[64][65];
    const int tid = threadIdx.x;
    const int n0 = blockIdx.x * 64;   // SDR index block
    const int h0 = blockIdx.y * 64;   // hidden index block

    #pragma unroll
    for (int i = 0; i < 16; ++i) {
        int idx = tid + i * 256;
        int r = idx >> 6;        // h offset
        int c = idx & 63;        // n offset
        t[r][c] = proj_w[(size_t)(h0 + r) * SDRN + (n0 + c)];
    }
    __syncthreads();
    #pragma unroll
    for (int i = 0; i < 16; ++i) {
        int idx = tid + i * 256;
        int r = idx >> 6;        // n offset
        int c = idx & 63;        // h offset
        Wt[(size_t)(n0 + r) * HID + (h0 + c)] = (_Float16)t[c][r];
    }
}

// ---------------------------------------------------------------------------
// Kernel 2: X = A(0/1) x Wt, f32 accum via WMMA.
//  - A materialized per K-step from a per-row bitmap (built once with ds_or).
//  - B staged pair-interleaved, fragment-major: each lane's frag = 2x b128.
//  - Next B tile prefetched into registers during the compute phase.
// ---------------------------------------------------------------------------
__global__ __launch_bounds__(256)
void sdr_gemm_kernel(const int* __restrict__ ids,
                     const _Float16* __restrict__ Wt,
                     float* __restrict__ xout) {
    // Row-occupancy bitmaps: 128 rows x 2048 bits, padded to kill bank conflicts
    __shared__ __align__(16) unsigned int sBit[MB][66];     // 33792 B
    // A tile, row-major [M][K], rows padded to 40 halves (80 B)
    __shared__ __align__(16) _Float16     sA[MB][40];       // 10240 B
    // B tile, fragment-major: sBf[n][kp] = packed {B[2kp][n], B[2kp+1][n]}
    __shared__ __align__(16) unsigned int sBf[NB][20];      // 10240 B

    const int tid  = threadIdx.x;
    const int pos0 = blockIdx.y * MB;
    const int h0   = blockIdx.x * NB;

    const int wave = tid >> 5;
    const int lane = tid & 31;
    const int wm   = wave & 3;    // M sub-block: rows [wm*32, wm*32+32)
    const int wn   = wave >> 2;   // N sub-block: cols [wn*64, wn*64+64)
    const int ln   = lane & 15;
    const int hi   = lane >> 4;

    // A-build ownership: thread owns 16 K-slots of one row
    const int arow  = tid >> 1;
    const int ahalf = tid & 1;            // k sub-range [16*ahalf, 16*ahalf+16)

    // --- Build bitmaps once (duplicates collapse via OR: set semantics) ---
    unsigned int* sBitu = &sBit[0][0];
    #pragma unroll
    for (int i = 0; i < (MB * 66) / 256; ++i)
        sBitu[tid + i * 256] = 0u;
    __syncthreads();
    for (int i = tid; i < MB * NW; i += 256) {
        int id  = ids[pos0 * NW + i];
        int row = i / NW;
        atomicOr(&sBit[row][id >> 5], 1u << (id & 31));
    }
    // visibility of bitmap ensured by the first loop barrier below

    unsigned int* sAu = (unsigned int*)&sA[0][0];   // row stride 20 dwords

    v8f acc[2][4] = {};

    // --- Prologue: prefetch B tile 0 into registers ---
    unsigned int pk[8];
    {
        const int k0 = 0;
        #pragma unroll
        for (int i = 0; i < 4; ++i) {
            int e  = tid + i * 256;
            int kp = e >> 6;
            int c  = (e & 63) * 2;
            const unsigned int u0 = *(const unsigned int*)(Wt + (size_t)(k0 + 2 * kp)     * HID + h0 + c);
            const unsigned int u1 = *(const unsigned int*)(Wt + (size_t)(k0 + 2 * kp + 1) * HID + h0 + c);
            pk[2 * i]     = (u0 & 0xffffu) | (u1 << 16);
            pk[2 * i + 1] = (u0 >> 16)     | (u1 & 0xffff0000u);
        }
    }

    for (int step = 0; step < KSTEPS; ++step) {
        __syncthreads();   // previous compute done reading sA/sBf (and bitmap ready)

        // --- Commit prefetched B tile to LDS ---
        #pragma unroll
        for (int i = 0; i < 4; ++i) {
            int e  = tid + i * 256;
            int kp = e >> 6;
            int c  = (e & 63) * 2;
            sBf[c][kp]     = pk[2 * i];
            sBf[c + 1][kp] = pk[2 * i + 1];
        }

        // --- Expand my 16 bitmap bits into 8 packed-f16 dwords of A ---
        {
            unsigned int w = sBit[arow][step] >> (16 * ahalf);
            FragU av;
            #pragma unroll
            for (int j = 0; j < 8; ++j) {
                unsigned int b = (w >> (2 * j)) & 3u;
                av.u[j] = ((b & 1u) ? 0x00003C00u : 0u) | ((b & 2u) ? 0x3C000000u : 0u);
            }
            v4u* rowp = (v4u*)&sAu[arow * 20 + ahalf * 8];
            rowp[0] = av.q[0];
            rowp[1] = av.q[1];
        }

        // --- Prefetch next B tile into registers (hidden behind compute) ---
        if (step + 1 < KSTEPS) {
            const int k0 = (step + 1) * KC;
            #pragma unroll
            for (int i = 0; i < 4; ++i) {
                int e  = tid + i * 256;
                int kp = e >> 6;
                int c  = (e & 63) * 2;
                const unsigned int u0 = *(const unsigned int*)(Wt + (size_t)(k0 + 2 * kp)     * HID + h0 + c);
                const unsigned int u1 = *(const unsigned int*)(Wt + (size_t)(k0 + 2 * kp + 1) * HID + h0 + c);
                pk[2 * i]     = (u0 & 0xffffu) | (u1 << 16);
                pk[2 * i + 1] = (u0 >> 16)     | (u1 & 0xffff0000u);
            }
        }

        __syncthreads();   // tiles ready

        // --- Fragments: lane k runs [8hi,8hi+8) + [16+8hi,24+8hi) -> ds_load_b128 pairs ---
        FragU af[2];
        #pragma unroll
        for (int mt = 0; mt < 2; ++mt) {
            int M  = wm * 32 + mt * 16 + ln;
            int rb = M * 20 + 4 * hi;
            af[mt].q[0] = *(const v4u*)&sAu[rb];
            af[mt].q[1] = *(const v4u*)&sAu[rb + 8];
        }

        #pragma unroll
        for (int nt = 0; nt < 4; ++nt) {
            FragU bf;
            int N = wn * 64 + nt * 16 + ln;
            bf.q[0] = *(const v4u*)&sBf[N][4 * hi];
            bf.q[1] = *(const v4u*)&sBf[N][4 * hi + 8];

            acc[0][nt] = __builtin_amdgcn_wmma_f32_16x16x32_f16(
                false, af[0].h, false, bf.h, (short)0, acc[0][nt], false, false);
            acc[1][nt] = __builtin_amdgcn_wmma_f32_16x16x32_f16(
                false, af[1].h, false, bf.h, (short)0, acc[1][nt], false, false);
        }
    }

    // --- Store f32 pre-norm X (16x16 f32 C/D layout: VGPR r -> M = r + 8*hi)
    #pragma unroll
    for (int mt = 0; mt < 2; ++mt) {
        #pragma unroll
        for (int nt = 0; nt < 4; ++nt) {
            int col = h0 + wn * 64 + nt * 16 + ln;
            #pragma unroll
            for (int r = 0; r < 8; ++r) {
                int row = pos0 + wm * 32 + mt * 16 + r + 8 * hi;
                xout[(size_t)row * HID + col] = acc[mt][nt][r];
            }
        }
    }
}

// ---------------------------------------------------------------------------
// Kernel 3: in-place LayerNorm over hidden dim (one block per position).
// ---------------------------------------------------------------------------
__global__ __launch_bounds__(256)
void sdr_layernorm_kernel(float* __restrict__ x,
                          const float* __restrict__ gamma,
                          const float* __restrict__ beta) {
    __shared__ float s1[256];
    __shared__ float s2[256];
    const int tid = threadIdx.x;
    float* xr = x + (size_t)blockIdx.x * HID;

    float v[8];
    float sum = 0.f, sq = 0.f;
    #pragma unroll
    for (int i = 0; i < 8; ++i) {
        float t = xr[tid + i * 256];
        v[i] = t;
        sum += t;
        sq  += t * t;
    }
    s1[tid] = sum;
    s2[tid] = sq;
    __syncthreads();
    #pragma unroll
    for (int off = 128; off > 0; off >>= 1) {
        if (tid < off) {
            s1[tid] += s1[tid + off];
            s2[tid] += s2[tid + off];
        }
        __syncthreads();
    }
    const float mu  = s1[0] * (1.0f / HID);
    const float var = s2[0] * (1.0f / HID) - mu * mu;
    const float rs  = rsqrtf(var + EPS);
    #pragma unroll
    for (int i = 0; i < 8; ++i) {
        int c = tid + i * 256;
        xr[c] = (v[i] - mu) * rs * gamma[c] + beta[c];
    }
}

// ---------------------------------------------------------------------------
extern "C" void kernel_launch(void* const* d_in, const int* in_sizes, int n_in,
                              void* d_out, int out_size, void* d_ws, size_t ws_size,
                              hipStream_t stream) {
    const int*   ids    = (const int*)d_in[0];     // input_ids [B,S,41]
    const float* proj_w = (const float*)d_in[1];   // [HID, SDRN] f32
    const float* gamma  = (const float*)d_in[2];   // [HID]
    const float* beta   = (const float*)d_in[3];   // [HID]
    float*       out    = (float*)d_out;           // [B,S,HID] f32
    _Float16*    Wt     = (_Float16*)d_ws;         // [SDRN, HID] f16 (8 MB)

    // 1) proj_w -> Wt (transpose + f16 convert)
    dim3 tgrid(SDRN / 64, HID / 64);
    sdr_transpose_kernel<<<tgrid, 256, 0, stream>>>(proj_w, Wt);

    // 2) WMMA GEMM: X = A(0/1) x Wt, written into d_out (pre-norm, f32)
    dim3 ggrid(HID / NB, BATCH_S / MB);
    sdr_gemm_kernel<<<ggrid, 256, 0, stream>>>(ids, Wt, out);

    // 3) in-place LayerNorm
    sdr_layernorm_kernel<<<BATCH_S, 256, 0, stream>>>(out, gamma, beta);
}
